// SelfAttention_11613591569038
// MI455X (gfx1250) — matrix operand order repaired and checked by reading, not compiled, not run on hardware
//
#include <hip/hip_runtime.h>
#include <hip/hip_bf16.h>

typedef __attribute__((ext_vector_type(8)))  _Float16 v8h;
typedef __attribute__((ext_vector_type(16))) _Float16 v16h;
typedef __attribute__((ext_vector_type(8)))  float    v8f;

#define B_  4
#define C_  128
#define N_  16384   // 16*32*32
#define M_  2048    // N/8
#define D1_ 32
#define D2_ 64

__device__ __forceinline__ v8h ld8(const _Float16* p) { return *(const v8h*)p; }
__device__ __forceinline__ v16h cat16(v8h lo, v8h hi) {
  return __builtin_shufflevector(lo, hi, 0,1,2,3,4,5,6,7,8,9,10,11,12,13,14,15);
}

// Single-instruction DPP16 butterfly steps: combine with the row-rotated value
// in one VOP2+DPP op (x = op(ror(x), x)). Rows of 16 lanes == our half-waves.
#define ROWMAX_STEP(x, R)                                                      \
  asm("v_max_num_f32_dpp %0, %0, %0 row_ror:" R                                \
      " row_mask:0xf bank_mask:0xf bound_ctrl:1" : "+v"(x))
#define ROWADD_STEP(x, R)                                                      \
  asm("v_add_f32_dpp %0, %0, %0 row_ror:" R                                    \
      " row_mask:0xf bank_mask:0xf bound_ctrl:1" : "+v"(x))

__device__ __forceinline__ float rowmax16(float x) {
  ROWMAX_STEP(x, "1"); ROWMAX_STEP(x, "2");
  ROWMAX_STEP(x, "4"); ROWMAX_STEP(x, "8");
  return x;
}
__device__ __forceinline__ float rowsum16(float x) {
  ROWADD_STEP(x, "1"); ROWADD_STEP(x, "2");
  ROWADD_STEP(x, "4"); ROWADD_STEP(x, "8");
  return x;
}

// ---------------- Stage 1a: theta = relu(wf @ x + bf) * log2(e), [B*N][32] f16
__global__ __launch_bounds__(256) void conv_theta_kernel(
    const float* __restrict__ x, const float* __restrict__ wf,
    const float* __restrict__ bf, _Float16* __restrict__ theta)
{
  __shared__ float w[D1_ * C_];
  for (int i = threadIdx.x; i < D1_ * C_; i += 256) w[i] = wf[i];
  __syncthreads();
  const int idx = blockIdx.x * 256 + threadIdx.x;   // b*N + n
  const int b = idx >> 14;
  const int n = idx & (N_ - 1);
  float acc[D1_];
  #pragma unroll
  for (int o = 0; o < D1_; ++o) acc[o] = bf[o];
  const float* xp = x + (size_t)b * C_ * N_ + n;
  for (int c = 0; c < C_; ++c) {
    const float xv = xp[(size_t)c * N_];
    #pragma unroll
    for (int o = 0; o < D1_; ++o) acc[o] = fmaf(w[o * C_ + c], xv, acc[o]);
  }
  _Float16* tp = theta + (size_t)idx * D1_;
  #pragma unroll
  for (int o = 0; o < D1_; ++o)
    tp[o] = (_Float16)(fmaxf(acc[o], 0.f) * 1.44269504088896340736f);
}

// ---------------- Stage 1b: pooled conv: max_{2x2x2} relu(W @ x + b)
// TRANSPOSED=false -> out [B*M][OUT]  (phi, B-operand friendly for scores)
// TRANSPOSED=true  -> out [B*OUT][M]  (g^T, B-operand friendly for P@G)
template <int OUT, bool TRANSPOSED>
__global__ __launch_bounds__(256) void conv_pool_kernel(
    const float* __restrict__ x, const float* __restrict__ wt,
    const float* __restrict__ bias, _Float16* __restrict__ outp)
{
  __shared__ float w[OUT * C_];
  for (int i = threadIdx.x; i < OUT * C_; i += 256) w[i] = wt[i];
  __syncthreads();
  const int idx = blockIdx.x * 256 + threadIdx.x;   // b*M + m
  const int b = idx >> 11;
  const int m = idx & (M_ - 1);
  const int wp = m & 15, hp = (m >> 4) & 15, tp = m >> 8;
  float best[OUT];
  #pragma unroll
  for (int o = 0; o < OUT; ++o) best[o] = -__builtin_inff();
  for (int p = 0; p < 8; ++p) {
    const int t = 2 * tp + (p >> 2), h = 2 * hp + ((p >> 1) & 1), ww = 2 * wp + (p & 1);
    const int n = (t << 10) + (h << 5) + ww;
    const float* xp = x + (size_t)b * C_ * N_ + n;
    float acc[OUT];
    #pragma unroll
    for (int o = 0; o < OUT; ++o) acc[o] = bias[o];
    for (int c = 0; c < C_; ++c) {
      const float xv = xp[(size_t)c * N_];
      #pragma unroll
      for (int o = 0; o < OUT; ++o) acc[o] = fmaf(w[o * C_ + c], xv, acc[o]);
    }
    #pragma unroll
    for (int o = 0; o < OUT; ++o) best[o] = fmaxf(best[o], acc[o]);
  }
  #pragma unroll
  for (int o = 0; o < OUT; ++o) {
    const float v = fmaxf(best[o], 0.f);
    if (TRANSPOSED) outp[((size_t)(b * OUT + o)) * M_ + m] = (_Float16)v;
    else            outp[(size_t)idx * OUT + o]            = (_Float16)v;
  }
}

// ---------------- Stage 2: flash attention. One wave owns 16 query rows.
// S(16x16) = theta(16x32) x phi^T(32x16) : one v_wmma_f32_16x16x32_f16 per key tile.
// Online softmax (log2-domain), then O += P(16x32) x G(32x64): 4 WMMAs per step.
__global__ __launch_bounds__(256) void fa_kernel(
    const _Float16* __restrict__ theta,
    const _Float16* __restrict__ phi,
    const _Float16* __restrict__ gt,
    float* __restrict__ attn_g)
{
  __shared__ _Float16 smem[8 * 16 * 32];          // 1KB P-tile per wave
  const int lane = threadIdx.x & 31;
  const int wv   = threadIdx.x >> 5;
  const int tile = blockIdx.x * 8 + wv;           // 0 .. B*N/16-1
  const int b    = tile >> 10;                    // N/16 = 1024 tiles per batch
  const int n0   = (tile & 1023) << 4;
  const int half = lane >> 4;                     // 0: lanes 0-15, 1: lanes 16-31
  const int l16  = lane & 15;

  // A operand (theta): lane row = n0+l16; elems 0..15 -> K=half*16 + i (contiguous)
  const _Float16* trow = theta + ((size_t)(b * N_ + n0 + l16)) * D1_;
  const v16h a_th = cat16(ld8(trow + half * 16), ld8(trow + half * 16 + 8));

  float mrow[8], lrow[8];
  v8f o0 = {}, o1 = {}, o2 = {}, o3 = {};
  #pragma unroll
  for (int j = 0; j < 8; ++j) { mrow[j] = -__builtin_inff(); lrow[j] = 0.f; }

  _Float16* pl = smem + wv * 512;                 // [16 rows][32 k] f16

  for (int m0 = 0; m0 < M_; m0 += 32) {
    // B operands (phi): col = key m0+l16, K = half*16 + i (contiguous channels)
    const _Float16* pr0 = phi + ((size_t)(b * M_ + m0 + l16)) * D1_ + half * 16;
    const _Float16* pr1 = pr0 + 16 * D1_;
    const v16h b_p0 = cat16(ld8(pr0), ld8(pr0 + 8));
    const v16h b_p1 = cat16(ld8(pr1), ld8(pr1 + 8));

    const v8f zero = {};
    v8f s0 = __builtin_amdgcn_wmma_f32_16x16x32_f16(false, a_th, false, b_p0,
                                                    (short)0, zero, false, false);
    v8f s1 = __builtin_amdgcn_wmma_f32_16x16x32_f16(false, a_th, false, b_p1,
                                                    (short)0, zero, false, false);

    // online softmax (log2 domain): row j+8*half lives across this 16-lane row
    float p0[8], p1[8];
    #pragma unroll
    for (int j = 0; j < 8; ++j) {
      const float tm = rowmax16(fmaxf(s0[j], s1[j]));
      const float nm = fmaxf(mrow[j], tm);
      const float sc = __builtin_exp2f(mrow[j] - nm);
      p0[j] = __builtin_exp2f(s0[j] - nm);
      p1[j] = __builtin_exp2f(s1[j] - nm);
      const float rs = rowsum16(p0[j] + p1[j]);
      lrow[j] = fmaf(lrow[j], sc, rs);
      mrow[j] = nm;
      o0[j] *= sc; o1[j] *= sc; o2[j] *= sc; o3[j] *= sc;
    }

    // C-layout -> A-layout transpose of P through per-wave LDS tile
    __builtin_amdgcn_wave_barrier();
    #pragma unroll
    for (int j = 0; j < 8; ++j) {
      const int r = j + 8 * half;
      pl[r * 32 + l16]      = (_Float16)p0[j];
      pl[r * 32 + 16 + l16] = (_Float16)p1[j];
    }
    __builtin_amdgcn_wave_barrier();
    asm volatile("s_wait_dscnt 0x0" ::: "memory");
    const _Float16* prw = pl + l16 * 32 + half * 16;
    const v16h a_p = cat16(*(const v8h*)prw, *(const v8h*)(prw + 8));
    __builtin_amdgcn_wave_barrier();

    // B operands from g^T: col = d2 channel t*16+l16, K = half*16 + i (contiguous m)
    const _Float16* g0 = gt + ((size_t)(b * D2_ + 0  + l16)) * M_ + m0 + half * 16;
    const _Float16* g1 = gt + ((size_t)(b * D2_ + 16 + l16)) * M_ + m0 + half * 16;
    const _Float16* g2 = gt + ((size_t)(b * D2_ + 32 + l16)) * M_ + m0 + half * 16;
    const _Float16* g3 = gt + ((size_t)(b * D2_ + 48 + l16)) * M_ + m0 + half * 16;
    o0 = __builtin_amdgcn_wmma_f32_16x16x32_f16(false, a_p, false, cat16(ld8(g0), ld8(g0 + 8)),
                                                (short)0, o0, false, false);
    o1 = __builtin_amdgcn_wmma_f32_16x16x32_f16(false, a_p, false, cat16(ld8(g1), ld8(g1 + 8)),
                                                (short)0, o1, false, false);
    o2 = __builtin_amdgcn_wmma_f32_16x16x32_f16(false, a_p, false, cat16(ld8(g2), ld8(g2 + 8)),
                                                (short)0, o2, false, false);
    o3 = __builtin_amdgcn_wmma_f32_16x16x32_f16(false, a_p, false, cat16(ld8(g3), ld8(g3 + 8)),
                                                (short)0, o3, false, false);
  }

  // normalize and write attn_g as [B*N][64] f32
  #pragma unroll
  for (int j = 0; j < 8; ++j) {
    const float inv = 1.f / lrow[j];
    const int r = j + 8 * half;
    const size_t base = ((size_t)(b * N_ + n0 + r)) * D2_ + l16;
    attn_g[base +  0] = o0[j] * inv;
    attn_g[base + 16] = o1[j] * inv;
    attn_g[base + 32] = o2[j] * inv;
    attn_g[base + 48] = o3[j] * inv;
  }
}

// ---------------- Stage 3: out = x + gamma * relu(wo @ attn_g + bo)
__global__ __launch_bounds__(256) void out_kernel(
    const float* __restrict__ x, const float* __restrict__ ag,
    const float* __restrict__ wo, const float* __restrict__ bo,
    const float* __restrict__ gamma, float* __restrict__ out)
{
  __shared__ float w[C_ * D2_];
  for (int i = threadIdx.x; i < C_ * D2_; i += 256) w[i] = wo[i];
  __syncthreads();
  const float gm = gamma[0];
  const int idx = blockIdx.x * 256 + threadIdx.x;   // b*N + n
  const int b = idx >> 14;
  const int n = idx & (N_ - 1);
  float a[D2_];
  const float* ap = ag + (size_t)idx * D2_;
  #pragma unroll
  for (int k = 0; k < D2_; ++k) a[k] = ap[k];
  for (int o = 0; o < C_; ++o) {
    float s = bo[o];
    #pragma unroll
    for (int k = 0; k < D2_; ++k) s = fmaf(w[o * D2_ + k], a[k], s);
    const size_t oo = ((size_t)(b * C_ + o)) * N_ + n;
    out[oo] = x[oo] + gm * fmaxf(s, 0.f);
  }
}

extern "C" void kernel_launch(void* const* d_in, const int* in_sizes, int n_in,
                              void* d_out, int out_size, void* d_ws, size_t ws_size,
                              hipStream_t stream) {
  (void)in_sizes; (void)n_in; (void)out_size; (void)ws_size;
  const float* x     = (const float*)d_in[0];
  const float* wf    = (const float*)d_in[1];
  const float* bf    = (const float*)d_in[2];
  const float* wg    = (const float*)d_in[3];
  const float* bg    = (const float*)d_in[4];
  const float* wh    = (const float*)d_in[5];
  const float* bh    = (const float*)d_in[6];
  const float* wo    = (const float*)d_in[7];
  const float* bo    = (const float*)d_in[8];
  const float* gamma = (const float*)d_in[9];
  float* out = (float*)d_out;

  char* ws = (char*)d_ws;
  _Float16* theta = (_Float16*)ws;                                       // 4 MB
  _Float16* phi   = (_Float16*)(ws + (size_t)(4u << 20));                // 512 KB
  _Float16* gt    = (_Float16*)(ws + (size_t)(4u << 20) + (512u << 10)); // 1 MB
  float*    ag    = (float*)(ws + (size_t)(4u << 20) + (512u << 10) + (1u << 20)); // 16 MB

  conv_theta_kernel<<<B_ * N_ / 256, 256, 0, stream>>>(x, wf, bf, theta);
  conv_pool_kernel<D1_, false><<<B_ * M_ / 256, 256, 0, stream>>>(x, wg, bg, phi);
  conv_pool_kernel<D2_, true ><<<B_ * M_ / 256, 256, 0, stream>>>(x, wh, bh, gt);
  fa_kernel<<<(B_ * N_ / 16) / 8, 256, 0, stream>>>(theta, phi, gt, ag);
  out_kernel<<<B_ * N_ / 256, 256, 0, stream>>>(x, ag, wo, bo, gamma, out);
}
